// CrossAttention_24034636988611
// MI455X (gfx1250) — matrix operand order repaired and checked by reading, not compiled, not run on hardware
//
#include <hip/hip_runtime.h>
#include <hip/hip_bf16.h>
#include <math.h>

// ---------------------------------------------------------------------------
// CDNA5 (gfx1250) cross-attention: all matmuls on v_wmma_f32_16x16x32_bf16,
// double-buffered LDS staging, packed b128 loads / b64 LDS stores.
// ---------------------------------------------------------------------------

typedef __attribute__((ext_vector_type(16))) __bf16 v16bf;
typedef __attribute__((ext_vector_type(8)))  float  v8f;

__device__ __forceinline__ unsigned short f2bf(float f) {
  union { __bf16 h; unsigned short u; } c;
  c.h = (__bf16)f;                 // native v_cvt on gfx1250
  return c.u;
}
__device__ __forceinline__ unsigned pack2bf(float a, float b) {
  return (unsigned)f2bf(a) | ((unsigned)f2bf(b) << 16);
}

union FragBF { v16bf v; unsigned u[8]; };

// A-fragment: 16x32 bf16, LDS tile row-major [m][k], stride in ushorts.
// ISA: lanes 0-15 -> M=0..15; VGPR r holds K = (r<4?0:16)+hh*8+(r%4)*2 (+1)
__device__ __forceinline__ v16bf load_frag_a(const unsigned short* lds, int stride,
                                             int m0, int k0) {
  FragBF f;
  int lane = threadIdx.x & 31;
  int m  = m0 + (lane & 15);
  int hh = lane >> 4;
  const unsigned short* row = lds + m * stride + k0;
#pragma unroll
  for (int r = 0; r < 8; ++r) {
    int k = ((r & 4) << 2) + hh * 8 + ((r & 3) << 1);
    f.u[r] = *(const unsigned*)(row + k);
  }
  return f.v;
}

// B-fragment: 32x16 bf16 from an LDS tile stored TRANSPOSED as [n][k].
// ISA: lane -> N = lane&15; VGPR r holds K = (r<4?0:16)+hh*8+(r%4)*2 (+1)
__device__ __forceinline__ v16bf load_frag_b(const unsigned short* lds, int stride,
                                             int n0, int k0) {
  FragBF f;
  int lane = threadIdx.x & 31;
  int n  = n0 + (lane & 15);
  int hh = lane >> 4;
  const unsigned short* row = lds + n * stride + k0;
#pragma unroll
  for (int r = 0; r < 8; ++r)
    f.u[r] = *(const unsigned*)(row + ((r & 4) << 2) + hh * 8 + ((r & 3) << 1));
  return f.v;
}

__device__ __forceinline__ v8f wmma_bf16(v16bf a, v16bf b, v8f c) {
  return __builtin_amdgcn_wmma_f32_16x16x32_bf16(false, a, false, b,
                                                 (short)0, c, false, false);
}

// ---------------------------------------------------------------------------
// Projection GEMM: out[M,N] = X[M,K] (f32) * W[K,N] (f32) + bias
// 256 threads = 8 waves (4 x 2); block tile 128x128, per-wave 32x64.
// Ping-pong LDS; 8 WMMA per 32-deep K-slab per wave.
// ---------------------------------------------------------------------------
#define PM 128
#define PN 128
#define PK 32
#define XSTR 36   // 18 dwords/row
#define WSTR 36

__device__ __forceinline__ void proj_stage(
    const float* __restrict__ X, const float* __restrict__ W,
    unsigned short* Xs, unsigned short* Wt,
    int row0, int col0, int kt, int K, int N, int tid) {
  // X tile 128x32: float4 loads, packed b64 LDS stores
#pragma unroll
  for (int it = 0; it < (PM * PK) / (256 * 4); ++it) {   // 4
    int i = tid + it * 256;
    int r = i >> 3, c4 = (i & 7) * 4;
    float4 v = *(const float4*)&X[(size_t)(row0 + r) * K + kt + c4];
    uint2 p; p.x = pack2bf(v.x, v.y); p.y = pack2bf(v.z, v.w);
    *(uint2*)&Xs[r * XSTR + c4] = p;
  }
  // W tile 32x128: float4 loads, transposed b16 stores into [n][k]
#pragma unroll
  for (int it = 0; it < (PK * PN) / (256 * 4); ++it) {   // 4
    int i = tid + it * 256;
    int kk = i >> 5, n4 = (i & 31) * 4;
    float4 v = *(const float4*)&W[(size_t)(kt + kk) * N + col0 + n4];
    Wt[(n4 + 0) * WSTR + kk] = f2bf(v.x);
    Wt[(n4 + 1) * WSTR + kk] = f2bf(v.y);
    Wt[(n4 + 2) * WSTR + kk] = f2bf(v.z);
    Wt[(n4 + 3) * WSTR + kk] = f2bf(v.w);
  }
}

template <bool STORE_BF16>
__global__ __launch_bounds__(256) void proj_gemm_kernel(
    const float* __restrict__ X, const float* __restrict__ W,
    const float* __restrict__ bias, unsigned short* __restrict__ out_bf,
    float* __restrict__ out_f, int M, int N, int K) {
  __shared__ unsigned short Xs[2][PM * XSTR];
  __shared__ unsigned short Wt[2][PN * WSTR];

  int tid  = threadIdx.x;
  int wave = tid >> 5;
  int wm   = wave & 3;      // 4 waves along M (32 rows each)
  int wn   = wave >> 2;     // 2 waves along N (64 cols each)
  int row0 = blockIdx.y * PM;
  int col0 = blockIdx.x * PN;

  v8f acc[2][4] = {};

  proj_stage(X, W, Xs[0], Wt[0], row0, col0, 0, K, N, tid);
  int cur = 0;
  for (int kt = 0; kt < K; kt += PK) {
    __syncthreads();
    if (kt + PK < K)
      proj_stage(X, W, Xs[cur ^ 1], Wt[cur ^ 1], row0, col0, kt + PK, K, N, tid);
    if (kt + 2 * PK < K)   // prefetch slab k+2 (global_prefetch_b8)
      __builtin_prefetch(&X[(size_t)(row0 + (tid >> 1)) * K + kt + 2 * PK + (tid & 1) * 16]);

    v16bf a0 = load_frag_a(Xs[cur], XSTR, wm * 32, 0);
    v16bf a1 = load_frag_a(Xs[cur], XSTR, wm * 32 + 16, 0);
#pragma unroll
    for (int tj = 0; tj < 4; ++tj) {
      v16bf b = load_frag_b(Wt[cur], WSTR, wn * 64 + tj * 16, 0);
      acc[0][tj] = wmma_bf16(a0, b, acc[0][tj]);
      acc[1][tj] = wmma_bf16(a1, b, acc[1][tj]);
    }
    cur ^= 1;
  }

  int lane = tid & 31;
  int hh = lane >> 4, ln = lane & 15;
#pragma unroll
  for (int ti = 0; ti < 2; ++ti)
#pragma unroll
    for (int tj = 0; tj < 4; ++tj) {
      int gcol = col0 + wn * 64 + tj * 16 + ln;
      float bv = bias[gcol];
#pragma unroll
      for (int r = 0; r < 8; ++r) {
        int grow = row0 + wm * 32 + ti * 16 + r + 8 * hh;
        float v = acc[ti][tj][r] + bv;
        if constexpr (STORE_BF16) out_bf[(size_t)grow * N + gcol] = f2bf(v);
        else                      out_f [(size_t)grow * N + gcol] = v;
      }
    }
}

// ---------------------------------------------------------------------------
// Flash attention: 128 threads = 4 waves; one WG per (b, h, 64-query tile).
// Ping-pong K/V staging; 16 WMMA per 64-key tile (8 QK^T + 8 PV).
// ---------------------------------------------------------------------------
#define QSTR 68   // 34 dwords/row, b64-aligned packed stores
#define KSTR 68
#define VSTR 68
#define PSTR 68

__device__ __forceinline__ void attn_stage_kv(
    const unsigned* __restrict__ ksrc, const unsigned* __restrict__ vsrc,
    unsigned short* Ks, unsigned short* Vt, int tid) {
  // K tile 64x64 bf16 ([key][d]): uint4 loads, two b64 stores each
#pragma unroll
  for (int it = 0; it < 4; ++it) {
    int i = tid + it * 128;
    int r = i >> 3, wq = (i & 7) * 4;
    uint4 v = *(const uint4*)&ksrc[(size_t)r * 512 + wq];
    uint2 p0; p0.x = v.x; p0.y = v.y;
    uint2 p1; p1.x = v.z; p1.y = v.w;
    *(uint2*)&Ks[r * KSTR + wq * 2]     = p0;
    *(uint2*)&Ks[r * KSTR + wq * 2 + 4] = p1;
  }
  // V tile 64x64 transposed into [d][kv]: uint2 load -> 4 b16 stores
#pragma unroll
  for (int it = 0; it < 8; ++it) {
    int i = tid + it * 128;
    int kv = i >> 4, wp = (i & 15) * 2;
    uint2 v = *(const uint2*)&vsrc[(size_t)kv * 512 + wp];
    int d0 = wp * 2;
    Vt[(d0 + 0) * VSTR + kv] = (unsigned short)v.x;
    Vt[(d0 + 1) * VSTR + kv] = (unsigned short)(v.x >> 16);
    Vt[(d0 + 2) * VSTR + kv] = (unsigned short)v.y;
    Vt[(d0 + 3) * VSTR + kv] = (unsigned short)(v.y >> 16);
  }
}

__global__ __launch_bounds__(128) void attention_kernel(
    const unsigned short* __restrict__ Q, const unsigned short* __restrict__ Kb,
    const unsigned short* __restrict__ Vb, float* __restrict__ O, int VSd) {
  __shared__ unsigned short Qs[64 * QSTR];         // [q][d]
  __shared__ unsigned short Ks[2][64 * KSTR];      // [key][d]
  __shared__ unsigned short Vt[2][64 * VSTR];      // [d][kv]
  __shared__ unsigned short Ps[4 * 16 * PSTR];     // per-wave P [q][kv]

  int bx = blockIdx.x;
  int lt = bx & 15;          // query tile
  int h  = (bx >> 4) & 15;   // head
  int b  = bx >> 8;          // batch
  int tid = threadIdx.x, wave = tid >> 5, lane = tid & 31;
  int hh = lane >> 4, ln = lane & 15;

  const int RS = 1024;
  size_t base = (size_t)b * 1024 * RS;
  int l0 = lt * 64;

  // Q tile 64x64 bf16: uint4 loads, b64 stores
  {
    const unsigned* src = (const unsigned*)(Q + base + (size_t)l0 * RS + h * 64);
#pragma unroll
    for (int it = 0; it < 4; ++it) {
      int i = tid + it * 128;
      int r = i >> 3, wq = (i & 7) * 4;
      uint4 v = *(const uint4*)&src[(size_t)r * 512 + wq];
      uint2 p0; p0.x = v.x; p0.y = v.y;
      uint2 p1; p1.x = v.z; p1.y = v.w;
      *(uint2*)&Qs[r * QSTR + wq * 2]     = p0;
      *(uint2*)&Qs[r * QSTR + wq * 2 + 4] = p1;
    }
  }

  const unsigned* kbase = (const unsigned*)(Kb + base + h * 64);
  const unsigned* vbase = (const unsigned*)(Vb + base + h * 64);

  float mrow[8], lrow[8];
  v8f o[4] = {};
#pragma unroll
  for (int r = 0; r < 8; ++r) { mrow[r] = -INFINITY; lrow[r] = 0.0f; }
  const float scale = 0.03125f;  // 1/sqrt(1024)

  attn_stage_kv(kbase, vbase, Ks[0], Vt[0], tid);
  int cur = 0;
  for (int kt0 = 0; kt0 < VSd; kt0 += 64) {
    __syncthreads();
    if (kt0 + 64 < VSd)
      attn_stage_kv(kbase + (size_t)(kt0 + 64) * 512,
                    vbase + (size_t)(kt0 + 64) * 512,
                    Ks[cur ^ 1], Vt[cur ^ 1], tid);
    if (kt0 + 128 < VSd) {   // prefetch tile k+2
      __builtin_prefetch(&kbase[(size_t)(kt0 + 128 + (tid >> 1)) * 512 + (tid & 1) * 16]);
      __builtin_prefetch(&vbase[(size_t)(kt0 + 128 + (tid >> 1)) * 512 + (tid & 1) * 16]);
    }

    const unsigned short* Kc = Ks[cur];
    const unsigned short* Vc = Vt[cur];

    // S = Q K^T : 16 rows x 64 keys, head dim 64 = 2 WMMA depths
    v16bf aq0 = load_frag_a(Qs, QSTR, wave * 16, 0);
    v16bf aq1 = load_frag_a(Qs, QSTR, wave * 16, 32);
    v8f s[4] = {};
#pragma unroll
    for (int g = 0; g < 4; ++g) {
      v16bf bk0 = load_frag_b(Kc, KSTR, g * 16, 0);
      v16bf bk1 = load_frag_b(Kc, KSTR, g * 16, 32);
      s[g] = wmma_bf16(aq0, bk0, s[g]);
      s[g] = wmma_bf16(aq1, bk1, s[g]);
    }

    // Online softmax (rows live in lane-halves of each C vgpr)
    unsigned short* Pw = Ps + wave * 16 * PSTR;
#pragma unroll
    for (int r = 0; r < 8; ++r) {
      float sv[4];
      float tmax = -INFINITY;
#pragma unroll
      for (int g = 0; g < 4; ++g) {
        sv[g] = s[g][r] * scale;
        tmax = fmaxf(tmax, sv[g]);
      }
      for (int off = 1; off < 16; off <<= 1)
        tmax = fmaxf(tmax, __shfl_xor(tmax, off, 32));
      float mnew  = fmaxf(mrow[r], tmax);
      float alpha = __expf(mrow[r] - mnew);
      float psum = 0.0f;
      int prow = r + 8 * hh;
#pragma unroll
      for (int g = 0; g < 4; ++g) {
        float p = __expf(sv[g] - mnew);
        psum += p;
        Pw[prow * PSTR + g * 16 + ln] = f2bf(p);
      }
      for (int off = 1; off < 16; off <<= 1)
        psum += __shfl_xor(psum, off, 32);
      lrow[r] = lrow[r] * alpha + psum;
      mrow[r] = mnew;
#pragma unroll
      for (int dg = 0; dg < 4; ++dg) o[dg][r] *= alpha;
    }

    // O += P V
    v16bf ap0 = load_frag_a(Pw, PSTR, 0, 0);
    v16bf ap1 = load_frag_a(Pw, PSTR, 0, 32);
#pragma unroll
    for (int dg = 0; dg < 4; ++dg) {
      v16bf bv0 = load_frag_b(Vc, VSTR, dg * 16, 0);
      v16bf bv1 = load_frag_b(Vc, VSTR, dg * 16, 32);
      o[dg] = wmma_bf16(ap0, bv0, o[dg]);
      o[dg] = wmma_bf16(ap1, bv1, o[dg]);
    }
    cur ^= 1;
  }

  // finalize: divide by row sums, write f32 [b][l][h*64+d]
#pragma unroll
  for (int r = 0; r < 8; ++r) {
    float inv = 1.0f / lrow[r];
    int l = l0 + wave * 16 + r + 8 * hh;
#pragma unroll
    for (int dg = 0; dg < 4; ++dg) {
      int d = h * 64 + dg * 16 + ln;
      O[base + (size_t)l * RS + d] = o[dg][r] * inv;
    }
  }
}

// ---------------------------------------------------------------------------
extern "C" void kernel_launch(void* const* d_in, const int* in_sizes, int n_in,
                              void* d_out, int out_size, void* d_ws, size_t ws_size,
                              hipStream_t stream) {
  const float* v_hs = (const float*)d_in[0];
  const float* l_hs = (const float*)d_in[1];
  const float* q_w  = (const float*)d_in[2];
  const float* q_b  = (const float*)d_in[3];
  const float* k_w  = (const float*)d_in[4];
  const float* k_b  = (const float*)d_in[5];
  const float* v_w  = (const float*)d_in[6];
  const float* v_b  = (const float*)d_in[7];
  const float* o_w  = (const float*)d_in[8];
  const float* o_b  = (const float*)d_in[9];
  float* out = (float*)d_out;

  const int Mtot = 4 * 1024, N = 1024, K = 1024;
  size_t qkv = (size_t)Mtot * N;                       // elements per QKV buffer
  unsigned short* Qbf = (unsigned short*)d_ws;         // 8 MB bf16
  unsigned short* Kbf = Qbf + qkv;                     // 8 MB
  unsigned short* Vbf = Kbf + qkv;                     // 8 MB
  float*          Att = (float*)(Vbf + qkv);           // 16 MB f32

  dim3 pgrid(N / PN, Mtot / PM);
  proj_gemm_kernel<true><<<pgrid, 256, 0, stream>>>(l_hs, q_w, q_b, Qbf, nullptr,
                                                    Mtot, N, K);
  proj_gemm_kernel<true><<<pgrid, 256, 0, stream>>>(v_hs, k_w, k_b, Kbf, nullptr,
                                                    Mtot, N, K);
  proj_gemm_kernel<true><<<pgrid, 256, 0, stream>>>(v_hs, v_w, v_b, Vbf, nullptr,
                                                    Mtot, N, K);
  attention_kernel<<<dim3(4 * 16 * 16), 128, 0, stream>>>(Qbf, Kbf, Vbf, Att, 1024);
  proj_gemm_kernel<false><<<pgrid, 256, 0, stream>>>(Att, o_w, o_b, nullptr, out,
                                                     Mtot, N, K);
}
